// CaptioningRNN_76682346103212
// MI455X (gfx1250) — matrix operand order repaired and checked by reading, not compiled, not run on hardware
//
#include <hip/hip_runtime.h>
#include <cstdint>
#include <cstddef>

typedef unsigned short ushort_t;
typedef __attribute__((ext_vector_type(16))) __bf16 v16bf;
typedef __attribute__((ext_vector_type(8)))  float  v8f;
typedef __attribute__((ext_vector_type(4)))  int    v4i;

#define NB   128
#define TT   32
#define DIN_ 1280
#define HD   1024
#define WVD  512
#define VV   10000
#define VVP  10048   // VV padded to multiple of 64 (N tile)
#define H4   4096
#define KCAT 2560
#define NLOC 16

#define MT   64      // GEMM M tile per block (4 x 16 per wave)
#define NT   64      // GEMM N tile per block (4 waves x 16)
#define KT   32      // GEMM K step
#define BPAD 8       // LDS row pad (bf16 elems) to rotate banks

#if defined(__AMDGCN__) && __has_builtin(__builtin_amdgcn_global_load_async_to_lds_b128)
#define HAVE_ASYNC_LDS 1
#else
#define HAVE_ASYNC_LDS 0
#endif
#if defined(__AMDGCN__) && __has_builtin(__builtin_amdgcn_s_wait_asynccnt)
#define HAVE_WAIT_ASYNC 1
#else
#define HAVE_WAIT_ASYNC 0
#endif

#define AS1 __attribute__((address_space(1)))
#define AS3 __attribute__((address_space(3)))

__device__ __forceinline__ ushort_t f2bf(float f) {
  unsigned u = __float_as_uint(f);
  unsigned r = u + 0x7FFFu + ((u >> 16) & 1u);   // round-to-nearest-even
  return (ushort_t)(r >> 16);
}

union FragAB { v16bf v; unsigned u[8]; v4i q[2]; };

// C(MxN) = A(MxK,bf16,row-major) * B(KxN,bf16,row-major) + bias[0..Nbias)
// Block: 128 threads = 4 waves. C tile: 64(M) x 64(N); wave w owns cols 16w..16w+15,
// and computes 4 M-subtiles reusing one B fragment per K-step (4 WMMA per frag build).
// B tiles (32x64 bf16, 4KB) double-buffered in LDS, staged with async-to-LDS so the
// next tile's loads overlap current-tile WMMAs (ASYNCcnt pipeline). A fragments load
// as 2x b128 from precomputed loop-invariant row pointers so all 8 loads clause ahead
// of the 4 back-to-back WMMAs.
// Requirements: M%64==0, N%64==0, K%32==0, rows 16B-aligned (2*lda,2*ldb % 16 == 0).
__global__ __launch_bounds__(128, 1) void wmma_gemm_bf16(
    const ushort_t* __restrict__ A, int lda,
    const ushort_t* __restrict__ B, int ldb,
    float* __restrict__ C, int ldc,
    const float* __restrict__ bias, int K, int Nbias)
{
  __shared__ alignas(16) ushort_t Bs[2][KT][NT + BPAD];

  const int tid  = threadIdx.x;
  const int lane = tid & 31;
  const int w    = tid >> 5;          // wave id 0..3
  const int r    = lane & 15;
  const int hs   = lane >> 4;

  const int mrow0 = blockIdx.y * MT;        // block's first row
  const int ncol0 = blockIdx.x * NT;        // block's first column
  const int cloc  = w * 16 + r;             // lane's column within tile
  const int col   = ncol0 + cloc;           // global column

  v8f acc[4];
  #pragma unroll
  for (int mi = 0; mi < 4; ++mi)
    #pragma unroll
    for (int i = 0; i < 8; ++i) acc[mi][i] = 0.0f;

  // loop-invariant per-lane A row pointers (element units); frag = b128 @ +0, +16B
  const ushort_t* aptr[4];
  #pragma unroll
  for (int mi = 0; mi < 4; ++mi)
    aptr[mi] = A + (size_t)(mrow0 + mi * 16 + r) * lda + 16 * hs;

  const int nk = K / KT;

  // stage B[kk..kk+31][ncol0..ncol0+63] into LDS buffer bufi (2 x 16B per thread)
  auto stage = [&](int kk, int bufi) {
    #pragma unroll
    for (int q = 0; q < 2; ++q) {
      int chunk = tid + q * 128;            // 0..255
      int kr    = chunk >> 3;               // tile row 0..31
      int cg    = chunk & 7;                // 8-col group
      const ushort_t* gp = B + (size_t)(kk + kr) * ldb + ncol0 + cg * 8;
      ushort_t* lp = &Bs[bufi][kr][cg * 8];
#if HAVE_ASYNC_LDS
      __builtin_amdgcn_global_load_async_to_lds_b128(
          (AS1 v4i*)(v4i*)(void*)const_cast<ushort_t*>(gp),
          (AS3 v4i*)(v4i*)(void*)lp, 0, 0);
#else
      const unsigned long long* g64 = (const unsigned long long*)gp;
      unsigned long long v0 = g64[0], v1 = g64[1];
      unsigned long long* l64 = (unsigned long long*)lp;
      l64[0] = v0; l64[1] = v1;
#endif
    }
  };

  stage(0, 0);   // prologue: tile 0 in flight

  for (int ik = 0; ik < nk; ++ik) {
    const int kk = ik * KT;

    // complete the tile staged for this iteration
#if HAVE_ASYNC_LDS
#if HAVE_WAIT_ASYNC
    __builtin_amdgcn_s_wait_asynccnt(0);
#else
    asm volatile("s_wait_asynccnt 0" ::: "memory");
#endif
#endif
    __syncthreads();   // all waves' staging visible; all prior reads of other buffer done

    // kick off next tile into the other buffer: overlaps the work below
    if (ik + 1 < nk) {
      stage(kk + KT, (ik + 1) & 1);
      __builtin_prefetch(aptr[0] + kk + KT, 0, 1);
    }

    // ---- all 4 A fragments: 8 x b128 loads, no per-iteration address math ----
    FragAB fa[4];
    #pragma unroll
    for (int mi = 0; mi < 4; ++mi) {
      const v4i* p = (const v4i*)(aptr[mi] + kk);
      fa[mi].q[0] = p[0];
      fa[mi].q[1] = p[1];   // +16 bytes (K offset +8 elements)
    }

    // ---- B 32x16 fragment from LDS (overlaps the global A loads above) ----
    const ushort_t (*Bt)[NT + BPAD] = Bs[ik & 1];
    FragAB fb;
    int kb = 16 * hs;
    #pragma unroll
    for (int i = 0; i < 8; ++i) {
      unsigned lo = Bt[kb + 2 * i][cloc];
      unsigned hi = Bt[kb + 2 * i + 1][cloc];
      fb.u[i] = lo | (hi << 16);
    }

    // ---- 4 back-to-back WMMAs on independent accumulators ----
    #pragma unroll
    for (int mi = 0; mi < 4; ++mi)
      acc[mi] = __builtin_amdgcn_wmma_f32_16x16x32_bf16(
          false, fa[mi].v, false, fb.v, (short)0, acc[mi], false, false);
  }

  float bv = (bias && col < Nbias) ? bias[col] : 0.0f;
  #pragma unroll
  for (int mi = 0; mi < 4; ++mi) {
    #pragma unroll
    for (int i = 0; i < 8; ++i) {
      int m = mrow0 + mi * 16 + 8 * hs + i;
      C[(size_t)m * ldc + col] = acc[mi][i] + bv;
    }
  }
}

__global__ void k_zero(float* p) { if (threadIdx.x == 0 && blockIdx.x == 0) p[0] = 0.f; }

__global__ void k_cvt(const float* __restrict__ s, ushort_t* __restrict__ d, int n) {
  int i = blockIdx.x * blockDim.x + threadIdx.x;
  if (i < n) d[i] = f2bf(s[i]);
}

// convert rows x cols f32 -> bf16 with padded leading dimension (pad cols zeroed)
__global__ void k_cvt_pad(const float* __restrict__ s, ushort_t* __restrict__ d,
                          int rows, int cols, int ldd) {
  int i = blockIdx.x * blockDim.x + threadIdx.x;
  if (i >= rows * ldd) return;
  int rr = i / ldd, cc = i % ldd;
  d[i] = (cc < cols) ? f2bf(s[(size_t)rr * cols + cc]) : (ushort_t)0;
}

// features (n, d, 4,4) -> featT[(n*16+l)][d] bf16
__global__ void k_featT(const float* __restrict__ f, ushort_t* __restrict__ o) {
  int i = blockIdx.x * blockDim.x + threadIdx.x;
  if (i >= NB * NLOC * DIN_) return;
  int d  = i % DIN_;
  int nl = i / DIN_;
  int n = nl / NLOC, l = nl % NLOC;
  o[i] = f2bf(f[((size_t)n * DIN_ + d) * NLOC + l]);
}

// x[n][t][:] = W_embed[captions[n][t]]  (t = 0..30), bf16
__global__ void k_embed(const int* __restrict__ cap, const float* __restrict__ We,
                        ushort_t* __restrict__ x) {
  int i = blockIdx.x * blockDim.x + threadIdx.x;
  if (i >= NB * (TT - 1) * WVD) return;
  int k  = i % WVD;
  int nt = i / WVD;
  int n = nt / (TT - 1), t = nt % (TT - 1);
  int tok = cap[n * TT + t];
  x[i] = f2bf(We[(size_t)tok * WVD + k]);
}

// h0 = c0 = mean over 16 spatial of Astore; also seed ph slot of Abuf (bf16)
__global__ void k_init(const float* __restrict__ Ast, float* __restrict__ h,
                       float* __restrict__ c, ushort_t* __restrict__ Abuf) {
  int i = blockIdx.x * blockDim.x + threadIdx.x;
  if (i >= NB * HD) return;
  int n = i / HD, j = i % HD;
  float s = 0.f;
  #pragma unroll
  for (int l = 0; l < NLOC; ++l) s += Ast[((size_t)n * NLOC + l) * HD + j];
  s *= (1.0f / 16.0f);
  h[i] = s; c[i] = s;
  Abuf[(size_t)n * KCAT + WVD + j] = f2bf(s);
}

// per-n attention: softmax over 16 spatial dots; writes attn (bf16) + copies xt into Abuf
__global__ __launch_bounds__(256) void k_attn(
    const float* __restrict__ h, const float* __restrict__ Ast,
    const ushort_t* __restrict__ Xbf, ushort_t* __restrict__ Abuf, int t)
{
  const int n = blockIdx.x;
  const int tid = threadIdx.x;
  __shared__ float e_sh[NLOC];
  __shared__ float aw_sh[NLOC];
  if (tid < NLOC) e_sh[tid] = 0.f;
  __syncthreads();

  float acc[NLOC];
  #pragma unroll
  for (int l = 0; l < NLOC; ++l) acc[l] = 0.f;
  const float* Abase = Ast + (size_t)n * NLOC * HD;
  for (int j = tid; j < HD; j += 256) {
    float hv = h[(size_t)n * HD + j];
    #pragma unroll
    for (int l = 0; l < NLOC; ++l) acc[l] += hv * Abase[(size_t)l * HD + j];
  }
  #pragma unroll
  for (int l = 0; l < NLOC; ++l) {
    float v = acc[l];
    #pragma unroll
    for (int off = 16; off > 0; off >>= 1) v += __shfl_xor(v, off, 32);
    if ((tid & 31) == 0) atomicAdd(&e_sh[l], v);
  }
  __syncthreads();
  if (tid == 0) {
    const float scale = 0.03125f;  // 1/sqrt(1024)
    float mx = -1e30f;
    #pragma unroll
    for (int l = 0; l < NLOC; ++l) { e_sh[l] *= scale; mx = fmaxf(mx, e_sh[l]); }
    float s = 0.f;
    #pragma unroll
    for (int l = 0; l < NLOC; ++l) { float ex = expf(e_sh[l] - mx); aw_sh[l] = ex; s += ex; }
    float inv = 1.0f / s;
    #pragma unroll
    for (int l = 0; l < NLOC; ++l) aw_sh[l] *= inv;
  }
  __syncthreads();
  for (int j = tid; j < HD; j += 256) {
    float s = 0.f;
    #pragma unroll
    for (int l = 0; l < NLOC; ++l) s += aw_sh[l] * Abase[(size_t)l * HD + j];
    Abuf[(size_t)n * KCAT + WVD + HD + j] = f2bf(s);
  }
  for (int k = tid; k < WVD; k += 256)
    Abuf[(size_t)n * KCAT + k] = Xbf[((size_t)n * (TT - 1) + t) * WVD + k];
}

// LSTM gates; updates c,h (f32) and ph slot of Abuf (bf16)
__global__ void k_gate(const float* __restrict__ pre, float* __restrict__ c,
                       float* __restrict__ h, ushort_t* __restrict__ Abuf) {
  int i = blockIdx.x * blockDim.x + threadIdx.x;
  if (i >= NB * HD) return;
  int n = i / HD, j = i % HD;
  const float* p = pre + (size_t)n * H4;
  float ai = p[j], af = p[HD + j], ao = p[2 * HD + j], ag = p[3 * HD + j];
  float si = 1.f / (1.f + expf(-ai));
  float sf = 1.f / (1.f + expf(-af));
  float so = 1.f / (1.f + expf(-ao));
  float cn = sf * c[i] + si * tanhf(ag);
  float hn = so * tanhf(cn);
  c[i] = cn; h[i] = hn;
  Abuf[(size_t)n * KCAT + WVD + j] = f2bf(hn);
}

// per-n log-softmax + masked NLL accumulation into loss (scores ldc = VVP)
__global__ __launch_bounds__(256) void k_nll(
    const float* __restrict__ S, const int* __restrict__ cap, int t,
    float* __restrict__ loss)
{
  const int n = blockIdx.x;
  const int tid = threadIdx.x;
  const float* row = S + (size_t)n * VVP;
  __shared__ float red[8];
  __shared__ float gmax_sh;

  float lm = -1e30f;
  for (int v = tid; v < VV; v += 256) lm = fmaxf(lm, row[v]);
  #pragma unroll
  for (int off = 16; off > 0; off >>= 1) lm = fmaxf(lm, __shfl_xor(lm, off, 32));
  if ((tid & 31) == 0) red[tid >> 5] = lm;
  __syncthreads();
  if (tid == 0) {
    float m = red[0];
    #pragma unroll
    for (int w = 1; w < 8; ++w) m = fmaxf(m, red[w]);
    gmax_sh = m;
  }
  __syncthreads();
  float gmax = gmax_sh;
  float ls = 0.f;
  for (int v = tid; v < VV; v += 256) ls += expf(row[v] - gmax);
  #pragma unroll
  for (int off = 16; off > 0; off >>= 1) ls += __shfl_xor(ls, off, 32);
  __syncthreads();
  if ((tid & 31) == 0) red[tid >> 5] = ls;
  __syncthreads();
  if (tid == 0) {
    float s = 0.f;
    #pragma unroll
    for (int w = 0; w < 8; ++w) s += red[w];
    float lse = gmax + logf(s);
    int tgt = cap[n * TT + t + 1];
    if (tgt != 0) atomicAdd(loss, (lse - row[tgt]) * (1.0f / 128.0f));
  }
}

extern "C" void kernel_launch(void* const* d_in, const int* in_sizes, int n_in,
                              void* d_out, int out_size, void* d_ws, size_t ws_size,
                              hipStream_t stream) {
  (void)in_sizes; (void)n_in; (void)out_size; (void)ws_size;
  const float* features = (const float*)d_in[0];
  const int*   captions = (const int*)d_in[1];
  const float* W_proj   = (const float*)d_in[2];
  const float* b_proj   = (const float*)d_in[3];
  const float* W_embed  = (const float*)d_in[4];
  const float* Wx       = (const float*)d_in[5];
  const float* Wh       = (const float*)d_in[6];
  const float* Wattn    = (const float*)d_in[7];
  const float* b        = (const float*)d_in[8];
  const float* W_vocab  = (const float*)d_in[9];
  const float* b_vocab  = (const float*)d_in[10];
  float* loss = (float*)d_out;

  char* ws = (char*)d_ws;
  size_t off = 0;
  auto take = [&](size_t bytes) -> char* {
    char* p = ws + off;
    off += (bytes + 255) & ~(size_t)255;
    return p;
  };
  ushort_t* featT  = (ushort_t*)take((size_t)NB * NLOC * DIN_ * 2);
  ushort_t* WprojB = (ushort_t*)take((size_t)DIN_ * HD * 2);
  ushort_t* Wcat   = (ushort_t*)take((size_t)KCAT * H4 * 2);
  ushort_t* WvocB  = (ushort_t*)take((size_t)HD * VVP * 2);   // padded to VVP cols
  ushort_t* Xbf    = (ushort_t*)take((size_t)NB * (TT - 1) * WVD * 2);
  float*    Astore = (float*)take((size_t)NB * NLOC * HD * 4);
  float*    hbuf   = (float*)take((size_t)NB * HD * 4);
  float*    cbuf   = (float*)take((size_t)NB * HD * 4);
  ushort_t* Abuf   = (ushort_t*)take((size_t)NB * KCAT * 2);  // [xt(512) | ph(1024) | attn(1024)]
  float*    pre    = (float*)take((size_t)NB * H4 * 4);
  float*    scores = (float*)take((size_t)NB * VVP * 4);      // padded ldc

  k_zero<<<1, 1, 0, stream>>>(loss);

  int c1 = DIN_ * HD;       k_cvt<<<(c1 + 255) / 256, 256, 0, stream>>>(W_proj, WprojB, c1);
  int cx = WVD * H4;        k_cvt<<<(cx + 255) / 256, 256, 0, stream>>>(Wx, Wcat, cx);
  int chh = HD * H4;        k_cvt<<<(chh + 255) / 256, 256, 0, stream>>>(Wh, Wcat + (size_t)WVD * H4, chh);
                            k_cvt<<<(chh + 255) / 256, 256, 0, stream>>>(Wattn, Wcat + (size_t)(WVD + HD) * H4, chh);
  int cvp = HD * VVP;       k_cvt_pad<<<(cvp + 255) / 256, 256, 0, stream>>>(W_vocab, WvocB, HD, VV, VVP);
  int cf = NB * NLOC * DIN_; k_featT<<<(cf + 255) / 256, 256, 0, stream>>>(features, featT);
  int ce = NB * (TT - 1) * WVD; k_embed<<<(ce + 255) / 256, 256, 0, stream>>>(captions, W_embed, Xbf);

  // Projection GEMM: (2048 x 1280) x (1280 x 1024) -> Astore (n,l,h)
  wmma_gemm_bf16<<<dim3(HD / NT, (NB * NLOC) / MT), 128, 0, stream>>>(
      featT, DIN_, WprojB, HD, Astore, HD, b_proj, DIN_, HD);

  k_init<<<(NB * HD + 255) / 256, 256, 0, stream>>>(Astore, hbuf, cbuf, Abuf);

  for (int t = 0; t < TT - 1; ++t) {
    k_attn<<<NB, 256, 0, stream>>>(hbuf, Astore, Xbf, Abuf, t);
    // pre = [xt|h|attn](128x2560) @ Wcat(2560x4096) + b
    wmma_gemm_bf16<<<dim3(H4 / NT, NB / MT), 128, 0, stream>>>(
        Abuf, KCAT, Wcat, H4, pre, H4, b, KCAT, H4);
    k_gate<<<(NB * HD + 255) / 256, 256, 0, stream>>>(pre, cbuf, hbuf, Abuf);
    // scores_t = h_t(128x1024) @ W_vocab(1024x10048 padded) + b_vocab
    wmma_gemm_bf16<<<dim3(VVP / NT, NB / MT), 128, 0, stream>>>(
        Abuf + WVD, KCAT, WvocB, VVP, scores, VVP, b_vocab, HD, VV);
    k_nll<<<NB, 256, 0, stream>>>(scores, captions, t, loss);
  }
}
